// NaViTEncoder_62766652063760
// MI455X (gfx1250) — matrix operand order repaired and checked by reading, not compiled, not run on hardware
//
#include <hip/hip_runtime.h>
#include <math.h>

// ---------------- problem constants ----------------
#define BB 4
#define LL 2048
#define DD 768
#define HH 12
#define DH 64
#define MLPD 3072
#define MROWS (BB * LL)          // 8192 token rows
#define EPSV 1e-5f

typedef __attribute__((ext_vector_type(16))) __bf16 v16bf;
typedef __attribute__((ext_vector_type(8)))  float  v8f;

#define WMMA_BF16(A_, B_, C_) \
  __builtin_amdgcn_wmma_f32_16x16x32_bf16(false, (A_), false, (B_), (short)0, (C_), false, false)

static __device__ __forceinline__ __bf16 f2bf(float f) { return (__bf16)f; }

// =====================================================================
// LayerNorm: one block (256 thr = 8 waves) per row of 768
// =====================================================================
__global__ __launch_bounds__(256)
void ln_kernel(const float* __restrict__ x, const float* __restrict__ g,
               const float* __restrict__ b, float* __restrict__ y)
{
  const int row = blockIdx.x;
  const float* xr = x + (size_t)row * DD;
  float*       yr = y + (size_t)row * DD;
  const int tid = threadIdx.x;

  float lv[3];
  float s = 0.f;
#pragma unroll
  for (int i = 0; i < 3; ++i) { lv[i] = xr[tid + i * 256]; s += lv[i]; }

  __shared__ float red1[8];
  __shared__ float red2[8];
#pragma unroll
  for (int off = 1; off < 32; off <<= 1) s += __shfl_xor(s, off, 32);
  if ((tid & 31) == 0) red1[tid >> 5] = s;
  __syncthreads();
  float tot = 0.f;
#pragma unroll
  for (int i = 0; i < 8; ++i) tot += red1[i];
  const float mean = tot * (1.0f / (float)DD);

  float vs = 0.f;
#pragma unroll
  for (int i = 0; i < 3; ++i) { float d = lv[i] - mean; vs += d * d; }
#pragma unroll
  for (int off = 1; off < 32; off <<= 1) vs += __shfl_xor(vs, off, 32);
  if ((tid & 31) == 0) red2[tid >> 5] = vs;
  __syncthreads();
  float vtot = 0.f;
#pragma unroll
  for (int i = 0; i < 8; ++i) vtot += red2[i];
  const float rinv = rsqrtf(vtot * (1.0f / (float)DD) + EPSV);

#pragma unroll
  for (int i = 0; i < 3; ++i) {
    const int idx = tid + i * 256;
    yr[idx] = (lv[i] - mean) * rinv * g[idx] + b[idx];
  }
}

// =====================================================================
// GEMM: C[M,N] = A[M,K] * W(+bias)(+resid)(+gelu)
//   wtrans=1 : W stored [N,K]  (einsum 'ed,d' form: QKV / O proj)
//   wtrans=0 : W stored [K,N]  (plain @ : MLP)
// Block: 256 thr = 8 waves, tile 128(M) x 64(N); wave computes 32x32.
// Double-buffered LDS panels; register-staged b128 loads overlap WMMA.
// =====================================================================
#define BM 128
#define BN 64
#define BK 32

__global__ __launch_bounds__(256)
void gemm_wmma(const float* __restrict__ A, const float* __restrict__ W,
               const float* __restrict__ bias, const float* __restrict__ resid,
               float* __restrict__ C, int M, int N, int K, int wtrans, int gelu)
{
  __shared__ __bf16 As[2][BM][BK + 4];
  __shared__ __bf16 Ws[2][BN][BK + 4];

  const int tid  = threadIdx.x;
  const int lane = tid & 31;
  const int wave = tid >> 5;       // 0..7
  const int wm   = wave & 3;       // M sub: 4 x 32 = 128
  const int wn   = wave >> 2;      // N sub: 2 x 32 = 64
  const int m0   = blockIdx.y * BM;
  const int n0   = blockIdx.x * BN;
  const int half = lane >> 4;

  // ---- staging helpers: load-all-then-store so global loads pipeline ----
  auto load_panel = [&](int kk, float4 (&ra)[4], float4 (&rw)[2]) {
#pragma unroll
    for (int j = 0; j < 4; ++j) {
      const int i = tid + j * 256;
      const int r = i >> 3, c = (i & 7) << 2;        // 8 float4 per 32-wide row
      ra[j] = *(const float4*)(A + (size_t)(m0 + r) * K + kk + c);
    }
    if (wtrans) {
#pragma unroll
      for (int j = 0; j < 2; ++j) {
        const int i = tid + j * 256;
        const int r = i >> 3, c = (i & 7) << 2;
        rw[j] = *(const float4*)(W + (size_t)(n0 + r) * K + kk + c);
      }
    } else {
#pragma unroll
      for (int j = 0; j < 2; ++j) {
        const int i = tid + j * 256;
        const int kr = i >> 4, nc = (i & 15) << 2;   // 16 float4 per 64-wide N row
        rw[j] = *(const float4*)(W + (size_t)(kk + kr) * N + n0 + nc);
      }
    }
  };
  auto store_panel = [&](int buf, const float4 (&ra)[4], const float4 (&rw)[2]) {
#pragma unroll
    for (int j = 0; j < 4; ++j) {
      const int i = tid + j * 256;
      const int r = i >> 3, c = (i & 7) << 2;
      As[buf][r][c + 0] = f2bf(ra[j].x); As[buf][r][c + 1] = f2bf(ra[j].y);
      As[buf][r][c + 2] = f2bf(ra[j].z); As[buf][r][c + 3] = f2bf(ra[j].w);
    }
    if (wtrans) {
#pragma unroll
      for (int j = 0; j < 2; ++j) {
        const int i = tid + j * 256;
        const int r = i >> 3, c = (i & 7) << 2;
        Ws[buf][r][c + 0] = f2bf(rw[j].x); Ws[buf][r][c + 1] = f2bf(rw[j].y);
        Ws[buf][r][c + 2] = f2bf(rw[j].z); Ws[buf][r][c + 3] = f2bf(rw[j].w);
      }
    } else {
#pragma unroll
      for (int j = 0; j < 2; ++j) {
        const int i = tid + j * 256;
        const int kr = i >> 4, nc = (i & 15) << 2;
        Ws[buf][nc + 0][kr] = f2bf(rw[j].x); Ws[buf][nc + 1][kr] = f2bf(rw[j].y);
        Ws[buf][nc + 2][kr] = f2bf(rw[j].z); Ws[buf][nc + 3][kr] = f2bf(rw[j].w);
      }
    }
  };

  v8f acc[2][2] = {{{}, {}}, {{}, {}}};

  // ---- prologue: panel 0 ----
  {
    float4 ra[4], rw[2];
    load_panel(0, ra, rw);
    store_panel(0, ra, rw);
  }
  __syncthreads();

  int cur = 0;
  for (int k0 = 0; k0 < K; k0 += BK) {
    const int kn = k0 + BK;
    const bool more = kn < K;
    float4 ra[4], rw[2];
    if (more) {
      load_panel(kn, ra, rw);                       // issue early, consume late
      __builtin_prefetch(A + (size_t)(m0 + (tid >> 1)) * K + kn + BK, 0, 1);
    }

    // ---- fragments from current panel ----
    v16bf af[2], bfv[2];
    const int kb = half * 8;
#pragma unroll
    for (int ms = 0; ms < 2; ++ms) {
      const int am = wm * 32 + ms * 16 + (lane & 15);
#pragma unroll
      for (int e = 0; e < 16; ++e)
        af[ms][e] = As[cur][am][kb + ((e >> 3) << 4) + (e & 7)];
    }
    const int kbb = half * 16;
#pragma unroll
    for (int ns = 0; ns < 2; ++ns) {
      const int bn = wn * 32 + ns * 16 + (lane & 15);
#pragma unroll
      for (int e = 0; e < 16; ++e)
        bfv[ns][e] = Ws[cur][bn][kb + ((e >> 3) << 4) + (e & 7)];
    }
    // ---- 4 WMMAs, A-frag reuse across N ----
#pragma unroll
    for (int ms = 0; ms < 2; ++ms)
#pragma unroll
      for (int ns = 0; ns < 2; ++ns)
        acc[ms][ns] = WMMA_BF16(af[ms], bfv[ns], acc[ms][ns]);

    if (more) store_panel(cur ^ 1, ra, rw);         // fill other buffer
    __syncthreads();
    cur ^= 1;
  }

  // ---- epilogue: rows r + half*8, cols lane%16 ----
#pragma unroll
  for (int ms = 0; ms < 2; ++ms) {
    const int rbase = m0 + wm * 32 + ms * 16 + (half << 3);
#pragma unroll
    for (int ns = 0; ns < 2; ++ns) {
      const int col = n0 + wn * 32 + ns * 16 + (lane & 15);
      const float bb = bias ? bias[col] : 0.f;
#pragma unroll
      for (int r = 0; r < 8; ++r) {
        const int row = rbase + r;
        float v = acc[ms][ns][r] + bb;
        if (resid) v += resid[(size_t)row * N + col];
        if (gelu)  v = 0.5f * v * (1.0f + erff(v * 0.70710678118654752f));
        C[(size_t)row * N + col] = v;
      }
    }
  }
}

// =====================================================================
// Flash attention: block = 4 waves sharing one (b,h); each wave owns a
// 16-query tile. K/V tiles (32 keys x 64 dims) double-buffered in LDS;
// V transposed so PV B-frags are contiguous LDS reads. One barrier per
// k-tile; P reshape is wave-private (LDS pipe is in-order per wave).
// =====================================================================
__global__ __launch_bounds__(128)
void attn_wmma(const float* __restrict__ qkv, const float* __restrict__ mask,
               float* __restrict__ out)
{
  __shared__ __bf16 Ks[2][32][DH + 4];   // [key][d]
  __shared__ __bf16 Vt[2][DH][36];       // [d][key] (transposed)
  __shared__ __bf16 P[4][16][34];        // per-wave P staging

  const int tid  = threadIdx.x;
  const int lane = tid & 31;
  const int wv   = tid >> 5;                  // wave 0..3
  const int bh   = blockIdx.x >> 5;           // 32 q-chunks per (b,h)
  const int qc   = blockIdx.x & 31;
  const int qt   = qc * 4 + wv;               // q tile 0..127
  const int b    = bh / HH;
  const int h    = bh % HH;
  const int half = lane >> 4;

  const size_t S3D = 3 * DD;
  const float* base = qkv + (size_t)b * LL * S3D;
  const float* Qp = base + (size_t)h * DH;
  const float* Kp = base + DD + (size_t)h * DH;
  const float* Vp = base + 2 * DD + (size_t)h * DH;

  auto load_tile = [&](int key0, float4 (&rk)[4], float4 (&rv)[4]) {
#pragma unroll
    for (int j = 0; j < 4; ++j) {
      const int i = tid + j * 128;
      const int key = i >> 4, c = (i & 15) << 2;   // 16 float4 per 64-wide row
      rk[j] = *(const float4*)(Kp + (size_t)(key0 + key) * S3D + c);
      rv[j] = *(const float4*)(Vp + (size_t)(key0 + key) * S3D + c);
    }
  };
  auto store_tile = [&](int buf, const float4 (&rk)[4], const float4 (&rv)[4]) {
#pragma unroll
    for (int j = 0; j < 4; ++j) {
      const int i = tid + j * 128;
      const int key = i >> 4, c = (i & 15) << 2;
      Ks[buf][key][c + 0] = f2bf(rk[j].x); Ks[buf][key][c + 1] = f2bf(rk[j].y);
      Ks[buf][key][c + 2] = f2bf(rk[j].z); Ks[buf][key][c + 3] = f2bf(rk[j].w);
      Vt[buf][c + 0][key] = f2bf(rv[j].x); Vt[buf][c + 1][key] = f2bf(rv[j].y);
      Vt[buf][c + 2][key] = f2bf(rv[j].z); Vt[buf][c + 3][key] = f2bf(rv[j].w);
    }
  };

  // Q fragments (DH=64 -> two 32-deep A frags), kept in registers
  v16bf aq0, aq1;
  {
    const int qrow = qt * 16 + (lane & 15);
    const int kb = half * 16;
#pragma unroll
    for (int e = 0; e < 16; ++e) {
      const int d = kb + e;
      aq0[e] = f2bf(Qp[(size_t)qrow * S3D + d]);
      aq1[e] = f2bf(Qp[(size_t)qrow * S3D + 32 + d]);
    }
  }

  v8f oacc[4] = {{}, {}, {}, {}};
  float mrun[8], lrun[8];
#pragma unroll
  for (int r = 0; r < 8; ++r) { mrun[r] = -3.0e38f; lrun[r] = 0.f; }
  const float scale = 0.125f;                 // 1/sqrt(64)
  const int rowl = half * 8;

  // ---- prologue: tile 0 ----
  {
    float4 rk[4], rv[4];
    load_tile(0, rk, rv);
    store_tile(0, rk, rv);
  }
  __syncthreads();

  int cur = 0;
  for (int kt = 0; kt < LL / 32; ++kt) {
    const int key0 = kt * 32;
    const bool more = (kt + 1) < (LL / 32);
    float4 rk[4], rv[4];
    if (more) load_tile(key0 + 32, rk, rv);   // issue early, consume late

    // ---- S = Q.K^T for 32 keys: two 16x16 f32 frags ----
    v8f s0 = {}, s1 = {};
    {
      const int db = half * 16;
      v16bf bk;
      const int kA = lane & 15;               // keys 0..15 of tile
#pragma unroll
      for (int e = 0; e < 16; ++e) bk[e] = Ks[cur][kA][db + e];
      s0 = WMMA_BF16(aq0, bk, s0);
#pragma unroll
      for (int e = 0; e < 16; ++e) bk[e] = Ks[cur][kA][32 + db + e];
      s0 = WMMA_BF16(aq1, bk, s0);
      const int kB = kA + 16;                 // keys 16..31
#pragma unroll
      for (int e = 0; e < 16; ++e) bk[e] = Ks[cur][kB][db + e];
      s1 = WMMA_BF16(aq0, bk, s1);
#pragma unroll
      for (int e = 0; e < 16; ++e) bk[e] = Ks[cur][kB][32 + db + e];
      s1 = WMMA_BF16(aq1, bk, s1);
    }

    // ---- online softmax (rows live in 16-lane halves) ----
    float p0[8], p1[8];
#pragma unroll
    for (int r = 0; r < 8; ++r) {
      const int rg = qt * 16 + rowl + r;
      const int c0 = key0 + (lane & 15);
      float v0 = s0[r] * scale + mask[(size_t)rg * LL + c0];
      float v1 = s1[r] * scale + mask[(size_t)rg * LL + c0 + 16];
      float mx = fmaxf(v0, v1);
#pragma unroll
      for (int off = 1; off < 16; off <<= 1) mx = fmaxf(mx, __shfl_xor(mx, off, 32));
      const float nm = fmaxf(mrun[r], mx);
      const float al = __expf(mrun[r] - nm);
      v0 = __expf(v0 - nm);
      v1 = __expf(v1 - nm);
      float sm = v0 + v1;
#pragma unroll
      for (int off = 1; off < 16; off <<= 1) sm += __shfl_xor(sm, off, 32);
      lrun[r] = lrun[r] * al + sm;
      mrun[r] = nm;
      p0[r] = v0; p1[r] = v1;
#pragma unroll
      for (int dt = 0; dt < 4; ++dt) oacc[dt][r] *= al;
    }

    // ---- reshape P (wave-private LDS; HW keeps DS ops in order) ----
#pragma unroll
    for (int r = 0; r < 8; ++r) {
      P[wv][rowl + r][lane & 15]        = f2bf(p0[r]);
      P[wv][rowl + r][16 + (lane & 15)] = f2bf(p1[r]);
    }
    asm volatile("" ::: "memory");            // keep store->load program order
    v16bf pa;
    {
      const int m = lane & 15;
      const int kb = half * 8;
#pragma unroll
      for (int e = 0; e < 16; ++e)
        pa[e] = P[wv][m][kb + ((e >> 3) << 4) + (e & 7)];
    }

    // ---- O += P.V : 4 d-tiles of 16 (contiguous Vt reads) ----
    {
      const int dcol = lane & 15;
#pragma unroll
      for (int dt = 0; dt < 4; ++dt) {
        v16bf bv;
#pragma unroll
        for (int e = 0; e < 16; ++e)
          bv[e] = Vt[cur][dt * 16 + dcol][half * 8 + ((e >> 3) << 4) + (e & 7)];
        oacc[dt] = WMMA_BF16(pa, bv, oacc[dt]);
      }
    }

    if (more) store_tile(cur ^ 1, rk, rv);
    __syncthreads();
    cur ^= 1;
  }

  // ---- normalize + write ----
  const int dcol = lane & 15;
#pragma unroll
  for (int r = 0; r < 8; ++r) {
    const float inv = 1.0f / lrun[r];
    const int row = qt * 16 + rowl + r;
    float* orow = out + ((size_t)b * LL + row) * DD + (size_t)h * DH;
#pragma unroll
    for (int dt = 0; dt < 4; ++dt)
      orow[dt * 16 + dcol] = oacc[dt][r] * inv;
  }
}

// =====================================================================
extern "C" void kernel_launch(void* const* d_in, const int* in_sizes, int n_in,
                              void* d_out, int out_size, void* d_ws, size_t ws_size,
                              hipStream_t stream)
{
  (void)in_sizes; (void)n_in; (void)out_size; (void)ws_size;

  const float* packed = (const float*)d_in[0];
  const float* amask  = (const float*)d_in[1];
  const float* w_qkv  = (const float*)d_in[2];
  const float* b_qkv  = (const float*)d_in[3];
  const float* w_o    = (const float*)d_in[4];
  const float* b_o    = (const float*)d_in[5];
  const float* ln1_g  = (const float*)d_in[6];
  const float* ln1_b  = (const float*)d_in[7];
  const float* ln2_g  = (const float*)d_in[8];
  const float* ln2_b  = (const float*)d_in[9];
  const float* w1     = (const float*)d_in[10];
  const float* b1     = (const float*)d_in[11];
  const float* w2     = (const float*)d_in[12];
  const float* b2     = (const float*)d_in[13];
  const float* ln_g   = (const float*)d_in[14];
  const float* ln_b   = (const float*)d_in[15];

  float* x    = (float*)d_ws;
  float* nbuf = x    + (size_t)MROWS * DD;
  float* qkv  = nbuf + (size_t)MROWS * DD;
  float* attn = qkv  + (size_t)MROWS * 3 * DD;
  float* hbuf = attn + (size_t)MROWS * DD;

  hipMemcpyAsync(x, packed, (size_t)MROWS * DD * sizeof(float),
                 hipMemcpyDeviceToDevice, stream);

  const dim3 lnGrid(MROWS), lnBlk(256);
  const dim3 gemmBlk(256);
  const dim3 gQKV(3 * DD / BN, MROWS / BM);
  const dim3 gO(DD / BN, MROWS / BM);
  const dim3 gM1(MLPD / BN, MROWS / BM);
  const dim3 gM2(DD / BN, MROWS / BM);
  const dim3 aGrid(BB * HH * 32), aBlk(128);

  for (int l = 0; l < 12; ++l) {
    ln_kernel<<<lnGrid, lnBlk, 0, stream>>>(x, ln1_g + l * DD, ln1_b + l * DD, nbuf);
    gemm_wmma<<<gQKV, gemmBlk, 0, stream>>>(nbuf, w_qkv + (size_t)l * 3 * DD * DD,
                                            b_qkv + (size_t)l * 3 * DD, nullptr, qkv,
                                            MROWS, 3 * DD, DD, /*wtrans=*/1, /*gelu=*/0);
    attn_wmma<<<aGrid, aBlk, 0, stream>>>(qkv, amask, attn);
    gemm_wmma<<<gO, gemmBlk, 0, stream>>>(attn, w_o + (size_t)l * DD * DD,
                                          b_o + (size_t)l * DD, x, x,
                                          MROWS, DD, DD, /*wtrans=*/1, /*gelu=*/0);
    ln_kernel<<<lnGrid, lnBlk, 0, stream>>>(x, ln2_g + l * DD, ln2_b + l * DD, nbuf);
    gemm_wmma<<<gM1, gemmBlk, 0, stream>>>(nbuf, w1 + (size_t)l * DD * MLPD,
                                           b1 + (size_t)l * MLPD, nullptr, hbuf,
                                           MROWS, MLPD, DD, /*wtrans=*/0, /*gelu=*/1);
    gemm_wmma<<<gM2, gemmBlk, 0, stream>>>(hbuf, w2 + (size_t)l * MLPD * DD,
                                           b2 + (size_t)l * DD, x, x,
                                           MROWS, DD, MLPD, /*wtrans=*/0, /*gelu=*/0);
  }
  ln_kernel<<<lnGrid, lnBlk, 0, stream>>>(x, ln_g, ln_b, (float*)d_out);
}
